// WindowAttention_46746424050481
// MI455X (gfx1250) — matrix operand order repaired and checked by reading, not compiled, not run on hardware
//
#include <hip/hip_runtime.h>

typedef __attribute__((ext_vector_type(16))) _Float16 v16h;
typedef __attribute__((ext_vector_type(8)))  _Float16 v8h;
typedef __attribute__((ext_vector_type(4)))  _Float16 v4h;
typedef __attribute__((ext_vector_type(8)))  float    v8f;
typedef __attribute__((ext_vector_type(4)))  float    v4f;

#define T_    64
#define C_    384
#define H_    12
#define HD_   32
#define NWIN  2048

// LDS strides (in elements)
#define XS 392   // xh / outh row stride (halves), 784B rows (16B aligned)
#define QS 48    // q row stride (halves), 96B (32B aligned)
#define KS 48    // k row stride (halves)
#define VS 80    // vT row stride (halves), 160B (32B aligned)
#define SS 65    // S row stride (floats)
#define PS 80    // P row stride (halves)

// LDS byte offsets
#define OFF_SF   0
#define OFF_XH   16640                       // 64*65*4
#define OFF_OUTH (OFF_XH + 50176)            // 64*392*2
#define OFF_QS   (OFF_OUTH + 50176)
#define OFF_KS   (OFF_QS + 6144)             // 64*48*2
#define OFF_VT   (OFF_KS + 6144)
#define OFF_PH   (OFF_VT + 5120)             // 32*80*2
#define LDS_BYTES (OFF_PH + 10240)           // => 144640 total

// d_ws byte offsets
#define WS_QKVW  0                            // 1152*384 f16 = 884736 B
#define WS_PROJW 884736                       // 384*384  f16 = 294912 B
#define WS_BIAS  (884736 + 294912)            // 12*64*64 f32 = 196608 B

static __device__ __forceinline__ v8f wmma32(v16h a, v16h b, v8f c) {
    // v_wmma_f32_16x16x32_f16: D = A(16x32) * B(32x16) + C(16x16 f32)
    return __builtin_amdgcn_wmma_f32_16x16x32_f16(
        /*neg_a=*/false, a, /*neg_b=*/false, b,
        /*c_mod=*/(short)0, c, /*reuse_a=*/false, /*reuse_b=*/false);
}

// A fragment (16x32 f16): lane (m = lane&15, half = lane>>4) holds
// K = [half*8, half*8+8) and [16+half*8, 16+half*8+8) of row m0+m.
static __device__ __forceinline__ v16h load_a(const _Float16* base, int ld,
                                              int m0, int k0, int lane) {
    int m = m0 + (lane & 15);
    int hf = lane >> 4;
    const _Float16* p = base + (size_t)m * ld + k0 + hf * 8;
    v8h lo = *(const v8h*)(p);
    v8h hi = *(const v8h*)(p + 16);
    v16h r;
#pragma unroll
    for (int i = 0; i < 8; ++i) { r[i] = lo[i]; r[i + 8] = hi[i]; }
    return r;
}

// B fragment (32x16 f16) from K-major storage BT[n][k] (stride ld halves):
// lane (n = lane&15, half = lane>>4) holds column n0+n,
// K = [half*8, half*8+8) and [16+half*8, 16+half*8+8).
static __device__ __forceinline__ v16h load_bt(const _Float16* base, int ld,
                                               int n0, int k0, int lane) {
    int n = n0 + (lane & 15);
    int hf = lane >> 4;
    const _Float16* p = base + (size_t)n * ld + k0 + hf * 8;
    v8h lo = *(const v8h*)(p);
    v8h hi = *(const v8h*)(p + 16);
    v16h r;
#pragma unroll
    for (int i = 0; i < 8; ++i) { r[i] = lo[i]; r[i + 8] = hi[i]; }
    return r;
}

// ---- pre-pass: weights -> f16, gather relative-position bias per head ----
__global__ void winattn_prep(const float* __restrict__ qkv_w,
                             const float* __restrict__ proj_w,
                             const float* __restrict__ rel_bias,
                             const int*   __restrict__ rel_idx,
                             _Float16* __restrict__ qkvw_h,
                             _Float16* __restrict__ projw_h,
                             float* __restrict__ bias_ws) {
    int i = blockIdx.x * blockDim.x + threadIdx.x;
    const int nq = 3 * C_ * C_;       // 442368
    const int np = C_ * C_;           // 147456
    const int nb = H_ * T_ * T_;      // 49152
    if (i < nq) qkvw_h[i] = (_Float16)qkv_w[i];
    if (i < np) projw_h[i] = (_Float16)proj_w[i];
    if (i < nb) {
        int h  = i / (T_ * T_);
        int ij = i % (T_ * T_);
        bias_ws[h * T_ * T_ + ij] = rel_bias[rel_idx[ij] * H_ + h];
    }
}

// ---- main fused window-attention kernel: one workgroup per window ----
__global__ __launch_bounds__(256) void winattn_main(
        const float* __restrict__ x,
        const float* __restrict__ qkv_b,
        const float* __restrict__ proj_b,
        const _Float16* __restrict__ qkvw_h,
        const _Float16* __restrict__ projw_h,
        const float* __restrict__ bias_ws,
        float* __restrict__ out) {
    extern __shared__ char smem[];
    float*    Sf   = (float*)   (smem + OFF_SF);
    _Float16* xh   = (_Float16*)(smem + OFF_XH);
    _Float16* outh = (_Float16*)(smem + OFF_OUTH);
    _Float16* qs   = (_Float16*)(smem + OFF_QS);
    _Float16* ks   = (_Float16*)(smem + OFF_KS);
    _Float16* vts  = (_Float16*)(smem + OFF_VT);
    _Float16* Ph   = (_Float16*)(smem + OFF_PH);

    const int b    = blockIdx.x;
    const int tid  = threadIdx.x;
    const int lane = tid & 31;
    const int w    = tid >> 5;
    const int nl   = lane & 15;
    const int mh   = (lane >> 4) * 8;
    const float scale = 0.17677669529663687f;  // 32^-0.5

    // stage x[b] into LDS as f16 (vectorized: float4 -> 4 halves)
    {
        const v4f* xb4 = (const v4f*)(x + (size_t)b * T_ * C_);
        for (int f = tid; f < T_ * C_ / 4; f += 256) {
            int r = f / (C_ / 4), c4 = f % (C_ / 4);
            v4f v = xb4[f];
            v4h hv;
#pragma unroll
            for (int i = 0; i < 4; ++i) hv[i] = (_Float16)v[i];
            *(v4h*)(xh + r * XS + c4 * 4) = hv;
        }
    }
    __syncthreads();

    for (int h = 0; h < H_; ++h) {
        // ---- per-head QKV projection: wave w owns tile (tm,tn) of q,k,v.
        // One A-fragment feeds 3 independent WMMA chains; fragment loads are
        // software-pipelined (double-buffered) so WMMAs never wait on the
        // load issued immediately before them.
        {
            int tm = w >> 1, tn = w & 1;
            int wrq = h * HD_ + tn * 16;           // q weight row base
            float bq = qkv_b[wrq + nl];
            float bk = qkv_b[C_ + wrq + nl];
            float bv = qkv_b[2 * C_ + wrq + nl];
            v8f aq = {bq, bq, bq, bq, bq, bq, bq, bq};
            v8f ak = {bk, bk, bk, bk, bk, bk, bk, bk};
            v8f av = {bv, bv, bv, bv, bv, bv, bv, bv};

            v16h a   = load_a(xh, XS, tm * 16, 0, lane);
            v16h bbq = load_bt(qkvw_h, C_, wrq, 0, lane);
            v16h bbk = load_bt(qkvw_h, C_, C_ + wrq, 0, lane);
            v16h bbv = load_bt(qkvw_h, C_, 2 * C_ + wrq, 0, lane);
#pragma unroll
            for (int k0 = 32; k0 < C_; k0 += 32) {
                v16h a_n   = load_a(xh, XS, tm * 16, k0, lane);
                v16h bbq_n = load_bt(qkvw_h, C_, wrq, k0, lane);
                v16h bbk_n = load_bt(qkvw_h, C_, C_ + wrq, k0, lane);
                v16h bbv_n = load_bt(qkvw_h, C_, 2 * C_ + wrq, k0, lane);
                aq = wmma32(a, bbq, aq);
                ak = wmma32(a, bbk, ak);
                av = wmma32(a, bbv, av);
                a = a_n; bbq = bbq_n; bbk = bbk_n; bbv = bbv_n;
            }
            aq = wmma32(a, bbq, aq);
            ak = wmma32(a, bbk, ak);
            av = wmma32(a, bbv, av);
#pragma unroll
            for (int r = 0; r < 8; ++r) {
                int m = tm * 16 + mh + r;
                qs[m * QS + tn * 16 + nl] = (_Float16)(aq[r] * scale);
                ks[m * QS + tn * 16 + nl] = (_Float16)ak[r];
                vts[(tn * 16 + nl) * VS + m] = (_Float16)av[r];  // vT[d][tok]
            }
        }
        __syncthreads();

        // ---- S = q @ k^T + rel_bias[h]: wave w owns tiles (tm0,tn),(tm0+2,tn)
        // sharing one B fragment; 2 independent WMMAs.
        const float* bh = bias_ws + (size_t)h * T_ * T_;
        {
            int tn = w & 3, tm0 = w >> 2;
            v16h bb = load_bt(ks, KS, tn * 16, 0, lane);
            v16h a0 = load_a(qs, QS, tm0 * 16, 0, lane);
            v16h a1 = load_a(qs, QS, (tm0 + 2) * 16, 0, lane);
            v8f s0 = {}, s1 = {};
            s0 = wmma32(a0, bb, s0);
            s1 = wmma32(a1, bb, s1);
            int n = tn * 16 + nl;
#pragma unroll
            for (int r = 0; r < 8; ++r) {
                int m0 = tm0 * 16 + mh + r;
                int m1 = m0 + 32;
                Sf[m0 * SS + n] = s0[r] + bh[m0 * T_ + n];
                Sf[m1 * SS + n] = s1[r] + bh[m1 * T_ + n];
            }
        }
        __syncthreads();

        // ---- row softmax: 4 threads per row, 16 cols each ----
        {
            int row  = tid >> 2;
            int part = tid & 3;
            const float* srow = Sf + row * SS + part * 16;
            float mx = -3.4e38f;
            float e[16];
#pragma unroll
            for (int i = 0; i < 16; ++i) mx = fmaxf(mx, srow[i]);
            mx = fmaxf(mx, __shfl_xor(mx, 1, 4));
            mx = fmaxf(mx, __shfl_xor(mx, 2, 4));
            float sum = 0.f;
#pragma unroll
            for (int i = 0; i < 16; ++i) { e[i] = __expf(srow[i] - mx); sum += e[i]; }
            sum += __shfl_xor(sum, 1, 4);
            sum += __shfl_xor(sum, 2, 4);
            float inv = 1.0f / sum;
            _Float16* prow = Ph + row * PS + part * 16;
#pragma unroll
            for (int i = 0; i < 16; ++i) prow[i] = (_Float16)(e[i] * inv);
        }
        __syncthreads();

        // ---- O_h = P @ V : 8 tiles (one per wave), K=64; all frags up front ----
        {
            int tm = w >> 1, tn = w & 1;
            v16h a0 = load_a(Ph, PS, tm * 16, 0, lane);
            v16h b0 = load_bt(vts, VS, tn * 16, 0, lane);
            v16h a1 = load_a(Ph, PS, tm * 16, 32, lane);
            v16h b1 = load_bt(vts, VS, tn * 16, 32, lane);
            v8f acc = {};
            acc = wmma32(a0, b0, acc);
            acc = wmma32(a1, b1, acc);
#pragma unroll
            for (int r = 0; r < 8; ++r)
                outh[(tm * 16 + mh + r) * XS + h * HD_ + tn * 16 + nl] =
                    (_Float16)acc[r];
        }
        __syncthreads();  // protect qs/ks/vts/Ph before next head
    }

    // ---- output projection: y = outh @ proj_w^T + proj_b ----
    // 96 tiles; pairs (j, j+8) as 2 interleaved, double-buffered WMMA chains.
    float* ob = out + (size_t)b * T_ * C_;
    for (int j = w; j < 96; j += 16) {
        int j2  = j + 8;
        int tm1 = j / 24,  tn1 = j % 24;
        int tm2 = j2 / 24, tn2 = j2 % 24;
        int n1 = tn1 * 16 + nl, n2 = tn2 * 16 + nl;
        float b1 = proj_b[n1], b2 = proj_b[n2];
        v8f acc1 = {b1, b1, b1, b1, b1, b1, b1, b1};
        v8f acc2 = {b2, b2, b2, b2, b2, b2, b2, b2};

        v16h a1 = load_a(outh, XS, tm1 * 16, 0, lane);
        v16h w1 = load_bt(projw_h, C_, tn1 * 16, 0, lane);
        v16h a2 = load_a(outh, XS, tm2 * 16, 0, lane);
        v16h w2 = load_bt(projw_h, C_, tn2 * 16, 0, lane);
#pragma unroll
        for (int k0 = 32; k0 < C_; k0 += 32) {
            v16h a1n = load_a(outh, XS, tm1 * 16, k0, lane);
            v16h w1n = load_bt(projw_h, C_, tn1 * 16, k0, lane);
            v16h a2n = load_a(outh, XS, tm2 * 16, k0, lane);
            v16h w2n = load_bt(projw_h, C_, tn2 * 16, k0, lane);
            acc1 = wmma32(a1, w1, acc1);
            acc2 = wmma32(a2, w2, acc2);
            a1 = a1n; w1 = w1n; a2 = a2n; w2 = w2n;
        }
        acc1 = wmma32(a1, w1, acc1);
        acc2 = wmma32(a2, w2, acc2);
#pragma unroll
        for (int r = 0; r < 8; ++r) {
            ob[(tm1 * 16 + mh + r) * C_ + n1] = acc1[r];
            ob[(tm2 * 16 + mh + r) * C_ + n2] = acc2[r];
        }
    }
}

extern "C" void kernel_launch(void* const* d_in, const int* in_sizes, int n_in,
                              void* d_out, int out_size, void* d_ws, size_t ws_size,
                              hipStream_t stream) {
    (void)in_sizes; (void)n_in; (void)out_size; (void)ws_size;
    const float* x        = (const float*)d_in[0];
    const float* qkv_w    = (const float*)d_in[1];
    const float* qkv_b    = (const float*)d_in[2];
    const float* proj_w   = (const float*)d_in[3];
    const float* proj_b   = (const float*)d_in[4];
    const float* rel_bias = (const float*)d_in[5];
    const int*   rel_idx  = (const int*)d_in[6];

    _Float16* qkvw_h  = (_Float16*)((char*)d_ws + WS_QKVW);
    _Float16* projw_h = (_Float16*)((char*)d_ws + WS_PROJW);
    float*    bias_ws = (float*)((char*)d_ws + WS_BIAS);

    int prep_n = 3 * C_ * C_;  // largest of the three prep ranges
    winattn_prep<<<(prep_n + 255) / 256, 256, 0, stream>>>(
        qkv_w, proj_w, rel_bias, rel_idx, qkvw_h, projw_h, bias_ws);

    (void)hipFuncSetAttribute((const void*)winattn_main,
                              hipFuncAttributeMaxDynamicSharedMemorySize,
                              LDS_BYTES);
    winattn_main<<<NWIN, 256, LDS_BYTES, stream>>>(
        x, qkv_b, proj_b, qkvw_h, projw_h, bias_ws, (float*)d_out);
}
